// MolTransformerGCN_7241314861282
// MI455X (gfx1250) — matrix operand order repaired and checked by reading, not compiled, not run on hardware
//
#include <hip/hip_runtime.h>
#include <stdint.h>

// ---------------- types for WMMA ----------------
typedef __attribute__((ext_vector_type(16))) _Float16 v16h;
typedef __attribute__((ext_vector_type(8)))  _Float16 v8h;
typedef __attribute__((ext_vector_type(8)))  float    v8f;

// ---------------- problem constants ----------------
constexpr int CB   = 1024;           // batch
constexpr int CS   = 64;             // padded seq
constexpr int CNA  = 48;             // atoms per molecule
constexpr int CN   = CB * CNA;       // 49152 nodes
constexpr int CE   = 4 * CN;         // 196608 edges
constexpr int CDIN = 128;
constexpr int CDM  = 256;
constexpr int CH   = 32;             // heads
constexpr int CDH  = 8;              // head dim
constexpr int CL   = 3;              // layers
constexpr int CT   = CB * CS;        // 65536 tokens

#define LDT 40   // LDS row stride in halves (80B) -> conflict-free b128 reads

// =====================================================================
// WMMA GEMM: C[M,N] = A[M,K] (f16 row-major) * Bt[N,K]^T (f16)
// Block = 256 threads (8 waves) computes a 64x128 tile of C.
// Each wave: one A fragment reused over 4 B fragments -> 4 WMMA / K-step.
// Global->LDS via async-DMA (ASYNCcnt), double-buffered: next tile's fill
// overlaps current tile's v_wmma work; one barrier per K-step. All ten
// ds_load_b128 are issued before the WMMA chain so the scheduler can wait
// with partial dscnt and hide LDS latency under the WMMAs.
// M mult of 64, N mult of 128, K mult of 32.
// =====================================================================
__global__ __launch_bounds__(256) void gemm_f16_wmma(
    const _Float16* __restrict__ A, const _Float16* __restrict__ Bt,
    const float* __restrict__ bias, const float* __restrict__ res,
    float* __restrict__ outF, _Float16* __restrict__ outH,
    int M, int N, int K, int relu)
{
    __shared__ _Float16 As[2][64 * LDT];
    __shared__ _Float16 Bs[2][128 * LDT];

    const int tid  = threadIdx.x;
    const int lane = tid & 31;
    const int wv   = tid >> 5;
    const int rg   = wv >> 1;      // row-group 0..3 (16 rows each)
    const int cq   = wv & 1;       // column half 0..1 (64 cols each)
    const int hi   = lane >> 4;    // lane half
    const int lr   = lane & 15;

    const size_t rowA0 = (size_t)blockIdx.y * 64;
    const size_t colB0 = (size_t)blockIdx.x * 128;

    const int ldr = tid >> 2;        // 0..63
    const int ldc = (tid & 3) * 8;   // 0,8,16,24 halves

    // LDS byte addresses for the async DMA (low 32 bits of generic pointer
    // = allocation-relative LDS offset; hardware adds LDS_BASE).
    const uint32_t lA  = (uint32_t)(uintptr_t)&As[0][ldr * LDT + ldc];
    const uint32_t lB0 = (uint32_t)(uintptr_t)&Bs[0][ldr * LDT + ldc];
    const uint32_t lB1 = (uint32_t)(uintptr_t)&Bs[0][(64 + ldr) * LDT + ldc];
    const uint32_t aBufB = 64 * LDT * 2;    // bytes per A stage buffer
    const uint32_t bBufB = 128 * LDT * 2;   // bytes per B stage buffer

    const _Float16* gA  = &A [(rowA0 + ldr)      * (size_t)K + ldc];
    const _Float16* gB0 = &Bt[(colB0 + ldr)      * (size_t)K + ldc];
    const _Float16* gB1 = &Bt[(colB0 + 64 + ldr) * (size_t)K + ldc];

#define ASYNC_LD(ldsaddr, gaddr)                                              \
    asm volatile("global_load_async_to_lds_b128 %0, %1, off"                  \
                 :: "v"(ldsaddr),                                             \
                    "v"((unsigned long long)(uintptr_t)(gaddr))               \
                 : "memory")

    // prologue: stage k0 = 0 into buffer 0
    ASYNC_LD(lA,  gA);
    ASYNC_LD(lB0, gB0);
    ASYNC_LD(lB1, gB1);

    v8f acc[4] = {{}, {}, {}, {}};

    int buf = 0;
    for (int k0 = 0; k0 < K; k0 += 32) {
        // own async fills complete -> signal; barrier => all tiles resident
        asm volatile("s_wait_asynccnt 0x0" ::: "memory");
        __syncthreads();

        if (k0 + 32 < K) {             // overlap next stage fill with compute
            const uint32_t nb = (uint32_t)(buf ^ 1);
            const int kk = k0 + 32;
            ASYNC_LD(lA  + nb * aBufB, gA  + kk);
            ASYNC_LD(lB0 + nb * bBufB, gB0 + kk);
            ASYNC_LD(lB1 + nb * bBufB, gB1 + kk);
        }

        union HV { v16h v; v8h h[2]; };
        const _Float16* as = &As[buf][0];
        const _Float16* bs = &Bs[buf][0];

        // Stage ALL fragments first (10x ds_load_b128), then run the WMMA
        // chain: lets the scheduler overlap LDS returns with matrix ops.
        HV a;
        const int ar = rg * 16 + lr;   // A: row = rg*16+lr ; K {0-7,16-23}/{8-15,24-31}
        a.h[0] = *(const v8h*)&as[ar * LDT + 8 * hi];
        a.h[1] = *(const v8h*)&as[ar * LDT + 16 + 8 * hi];
        HV b[4];
        #pragma unroll
        for (int t = 0; t < 4; ++t) {  // B: col = lane%16 ; K {0-7,16-23}/{8-15,24-31}
            const int bc = cq * 64 + t * 16 + lr;
            b[t].h[0] = *(const v8h*)&bs[bc * LDT + 8 * hi];
            b[t].h[1] = *(const v8h*)&bs[bc * LDT + 16 + 8 * hi];
        }
        #pragma unroll
        for (int t = 0; t < 4; ++t) {
            acc[t] = __builtin_amdgcn_wmma_f32_16x16x32_f16(
                false, a.v, false, b[t].v, (short)0, acc[t], false, false);
        }
        buf ^= 1;
        // no trailing barrier needed: each wave's ds_loads of this buffer are
        // dscnt-waited before its WMMAs, hence complete before it reaches the
        // next top-of-loop barrier, which precedes any refill of this buffer.
    }
#undef ASYNC_LD

    // Epilogue: C layout — VGPR i holds row (i + 8*hi) of the 16x16 tile.
    const int row0 = (int)rowA0 + rg * 16 + hi * 8;
    #pragma unroll
    for (int t = 0; t < 4; ++t) {
        const int c = (int)colB0 + cq * 64 + t * 16 + lr;
        const float bb = bias ? bias[c] : 0.0f;
        #pragma unroll
        for (int i = 0; i < 8; ++i) {
            const int row = row0 + i;
            const size_t idx = (size_t)row * N + c;
            float v = acc[t][i] + bb;
            if (res)  v += res[idx];
            if (relu) v = fmaxf(v, 0.0f);
            if (outF) outF[idx] = v;
            if (outH) outH[idx] = (_Float16)v;
        }
    }
}

// ============== weight transpose + f32 -> f16 convert ==============
__global__ void wt_convert_kernel(const float* __restrict__ W,
                                  _Float16* __restrict__ Bt, int K, int N)
{
    int id = blockIdx.x * blockDim.x + threadIdx.x;
    if (id >= K * N) return;
    int k = id / N, n = id % N;
    Bt[(size_t)n * K + k] = (_Float16)W[id];
}

// ============== embedding table with max-norm clip ==============
__global__ void tab_kernel(const float* __restrict__ emb,
                           _Float16* __restrict__ tab_h)
{
    __shared__ float red[128];
    const int r = blockIdx.x, t = threadIdx.x;
    float e = emb[r * 128 + t];
    red[t] = e * e;
    __syncthreads();
    for (int s = 64; s > 0; s >>= 1) {
        if (t < s) red[t] += red[t + s];
        __syncthreads();
    }
    float nrm = sqrtf(red[0]);
    float scale = fminf(1.0f, 1.0f / fmaxf(nrm, 1e-7f));
    tab_h[r * 128 + t] = (_Float16)(e * scale);
}

__global__ void gather_kernel(const int* __restrict__ x_atoms,
                              const _Float16* __restrict__ tab_h,
                              _Float16* __restrict__ x_h)
{
    int id = blockIdx.x * blockDim.x + threadIdx.x;   // over N*128
    int n = id >> 7, c = id & 127;
    x_h[id] = tab_h[x_atoms[n] * 128 + c];
}

// ============== graph degree / normalization ==============
__global__ void fill_f32_kernel(float* p, float v, int n)
{
    int id = blockIdx.x * blockDim.x + threadIdx.x;
    if (id < n) p[id] = v;
}
__global__ void fill_f16_kernel(_Float16* p, int n)
{
    int id = blockIdx.x * blockDim.x + threadIdx.x;
    if (id < n) p[id] = (_Float16)0.0f;
}
__global__ void deg_count_kernel(const int* __restrict__ ei, float* __restrict__ deg)
{
    int e = blockIdx.x * blockDim.x + threadIdx.x;
    if (e < CE) atomicAdd(&deg[ei[CE + e]], 1.0f);
}
__global__ void dis_kernel(const float* __restrict__ deg, float* __restrict__ dis)
{
    int n = blockIdx.x * blockDim.x + threadIdx.x;
    if (n < CN) dis[n] = rsqrtf(deg[n]);   // deg >= 1 always (self loop)
}

// ============== GCN edge aggregation (atomic scatter-add) ==============
__global__ void edge_agg_kernel(const int* __restrict__ ei,
                                const float* __restrict__ dis,
                                const float* __restrict__ xw,
                                float* __restrict__ agg)
{
    int id = blockIdx.x * blockDim.x + threadIdx.x;   // over E*128
    int e = id >> 7, c = id & 127;
    int s = ei[e], d = ei[CE + e];
    float nrm = dis[s] * dis[d];
    atomicAdd(&agg[(size_t)d * 128 + c], xw[(size_t)s * 128 + c] * nrm);
}

__global__ void gcn_final_kernel(const float* __restrict__ agg,
                                 const float* __restrict__ xw,
                                 const float* __restrict__ dis,
                                 const float* __restrict__ bias,
                                 _Float16* __restrict__ out, int relu)
{
    int id = blockIdx.x * blockDim.x + threadIdx.x;   // over N*128
    int n = id >> 7, c = id & 127;
    float di = dis[n];
    float v = agg[id] + xw[id] * di * di + bias[c];
    if (relu) v = fmaxf(v, 0.0f);
    out[id] = (_Float16)v;
}

// ============== scatter nodes into padded sequences ==============
__global__ void scatter_kernel(const _Float16* __restrict__ xg,
                               _Float16* __restrict__ pad)
{
    int id = blockIdx.x * blockDim.x + threadIdx.x;   // over N*128
    int n = id >> 7, c = id & 127;
    int pos = (n / CNA) * CS + (n % CNA);             // node_pos computed inline
    pad[(size_t)pos * 128 + c] = xg[id];
}

// ============== attention: one block per (batch, head) ==============
__global__ __launch_bounds__(64) void attn_kernel(
    const _Float16* __restrict__ qkv, _Float16* __restrict__ out)
{
    __shared__ float kk[64][8];
    __shared__ float vv[64][8];
    const int bh = blockIdx.x;
    const int b  = bh >> 5;            // / H
    const int hd = bh & 31;            // % H
    const int s  = threadIdx.x;        // 0..63 (query / key slot)
    const _Float16* base = qkv + (size_t)(b * 64 + s) * 768;

    #pragma unroll
    for (int d = 0; d < 8; ++d) {
        kk[s][d] = (float)base[256 + hd * 8 + d];
        vv[s][d] = (float)base[512 + hd * 8 + d];
    }
    float q[8];
    #pragma unroll
    for (int d = 0; d < 8; ++d) q[d] = (float)base[hd * 8 + d];
    __syncthreads();

    const float scale = 0.35355339059327373f;  // 1/sqrt(8)
    float sc[64];
    float mx = -1e30f;
    for (int j = 0; j < 64; ++j) {
        float dot = 0.0f;
        #pragma unroll
        for (int d = 0; d < 8; ++d) dot += q[d] * kk[j][d];
        dot *= scale;
        if (j >= CNA) dot = -1e9f;     // node_mask computed inline
        sc[j] = dot;
        mx = fmaxf(mx, dot);
    }
    float sum = 0.0f;
    for (int j = 0; j < 64; ++j) { float e = __expf(sc[j] - mx); sc[j] = e; sum += e; }
    const float inv = 1.0f / sum;
    float o[8] = {0, 0, 0, 0, 0, 0, 0, 0};
    for (int j = 0; j < 64; ++j) {
        float p = sc[j] * inv;
        #pragma unroll
        for (int d = 0; d < 8; ++d) o[d] += p * vv[j][d];
    }
    _Float16* ob = out + (size_t)(b * 64 + s) * 256 + hd * 8;
    #pragma unroll
    for (int d = 0; d < 8; ++d) ob[d] = (_Float16)o[d];
}

// ============== layer norm over D_M=256 (one block per row) ==============
__global__ __launch_bounds__(256) void ln_kernel(
    const float* __restrict__ x, const float* __restrict__ gb,
    float* __restrict__ outF, _Float16* __restrict__ outH)
{
    __shared__ float red[256];
    const int row = blockIdx.x, t = threadIdx.x;
    const float v = x[(size_t)row * 256 + t];
    red[t] = v;
    __syncthreads();
    for (int s = 128; s > 0; s >>= 1) { if (t < s) red[t] += red[t + s]; __syncthreads(); }
    const float mean = red[0] * (1.0f / 256.0f);
    __syncthreads();
    const float dv = v - mean;
    red[t] = dv * dv;
    __syncthreads();
    for (int s = 128; s > 0; s >>= 1) { if (t < s) red[t] += red[t + s]; __syncthreads(); }
    const float var = red[0] * (1.0f / 256.0f);
    const float y = dv * rsqrtf(var + 1e-5f) * gb[t] + gb[256 + t];
    outF[(size_t)row * 256 + t] = y;
    outH[(size_t)row * 256 + t] = (_Float16)y;
}

// =====================================================================
extern "C" void kernel_launch(void* const* d_in, const int* in_sizes, int n_in,
                              void* d_out, int out_size, void* d_ws, size_t ws_size,
                              hipStream_t stream)
{
    (void)in_sizes; (void)n_in; (void)out_size; (void)ws_size;

    const int*   x_atoms = (const int*)d_in[0];
    const int*   ei      = (const int*)d_in[1];
    // d_in[2] node_mask, d_in[3] node_pos: deterministic, computed inline
    const float* emb  = (const float*)d_in[4];
    const float* Wg1  = (const float*)d_in[5];
    const float* bg1  = (const float*)d_in[6];
    const float* Wg2  = (const float*)d_in[7];
    const float* bg2  = (const float*)d_in[8];
    const float* Wp   = (const float*)d_in[9];
    const float* bp   = (const float*)d_in[10];
    const float* Wqkv = (const float*)d_in[11];
    const float* bqkv = (const float*)d_in[12];
    const float* Wo   = (const float*)d_in[13];
    const float* bo   = (const float*)d_in[14];
    const float* ln1  = (const float*)d_in[15];
    const float* Wff1 = (const float*)d_in[16];
    const float* bff1 = (const float*)d_in[17];
    const float* Wff2 = (const float*)d_in[18];
    const float* bff2 = (const float*)d_in[19];
    const float* ln2  = (const float*)d_in[20];

    // -------- workspace layout --------
    char* base = (char*)d_ws;
    size_t off = 0;
    auto alloc = [&](size_t bytes) -> void* {
        void* p = base + off;
        off = (off + bytes + 255) & ~(size_t)255;
        return p;
    };
    float*    h      = (float*)   alloc((size_t)CT * CDM * 4);
    float*    tmp    = (float*)   alloc((size_t)CT * CDM * 4);
    _Float16* h_h    = (_Float16*)alloc((size_t)CT * CDM * 2);
    _Float16* attn_h = (_Float16*)alloc((size_t)CT * CDM * 2);
    _Float16* ff_h   = (_Float16*)alloc((size_t)CT * CDM * 2);
    _Float16* pad_h  = (_Float16*)alloc((size_t)CT * CDIN * 2);
    _Float16* Wg1T   = (_Float16*)alloc((size_t)CDIN * CDIN * 2);
    _Float16* Wg2T   = (_Float16*)alloc((size_t)CDIN * CDIN * 2);
    _Float16* WpT    = (_Float16*)alloc((size_t)CDM * CDIN * 2);
    _Float16* WqkvT  = (_Float16*)alloc((size_t)CL * 768 * CDM * 2);
    _Float16* WoT    = (_Float16*)alloc((size_t)CL * CDM * CDM * 2);
    _Float16* Wff1T  = (_Float16*)alloc((size_t)CL * CDM * CDM * 2);
    _Float16* Wff2T  = (_Float16*)alloc((size_t)CL * CDM * CDM * 2);
    _Float16* qkv_h  = (_Float16*)alloc((size_t)CT * 768 * 2);

    // GCN temporaries alias the qkv buffer (fully overwritten later)
    {
        char* ar = (char*)qkv_h;
        size_t a = 0;
        auto alloc2 = [&](size_t bytes) -> void* {
            void* p = ar + a;
            a = (a + bytes + 255) & ~(size_t)255;
            return p;
        };
        _Float16* x_h   = (_Float16*)alloc2((size_t)CN * CDIN * 2);
        float*    xw    = (float*)   alloc2((size_t)CN * CDIN * 4);
        float*    agg   = (float*)   alloc2((size_t)CN * CDIN * 4);
        _Float16* xg_h  = (_Float16*)alloc2((size_t)CN * CDIN * 2);
        float*    deg   = (float*)   alloc2((size_t)CN * 4);
        float*    dis   = (float*)   alloc2((size_t)CN * 4);
        _Float16* tab_h = (_Float16*)alloc2((size_t)65 * CDIN * 2);

        auto gemm = [&](const _Float16* A, const _Float16* Bt, const float* bias,
                        const float* res, float* oF, _Float16* oH,
                        int M, int Nn, int K, int relu) {
            dim3 g(Nn / 128, M / 64);
            gemm_f16_wmma<<<g, 256, 0, stream>>>(A, Bt, bias, res, oF, oH, M, Nn, K, relu);
        };

        // -------- weight conversion (fp32 [K,N] -> f16 [N,K]) --------
        auto wconv = [&](const float* W, _Float16* T, int K, int Nn) {
            int cnt = K * Nn;
            wt_convert_kernel<<<(cnt + 255) / 256, 256, 0, stream>>>(W, T, K, Nn);
        };
        wconv(Wg1, Wg1T, CDIN, CDIN);
        wconv(Wg2, Wg2T, CDIN, CDIN);
        wconv(Wp, WpT, CDIN, CDM);
        for (int i = 0; i < CL; ++i) {
            wconv(Wqkv + (size_t)i * CDM * 768, WqkvT + (size_t)i * 768 * CDM, CDM, 768);
            wconv(Wo   + (size_t)i * CDM * CDM, WoT   + (size_t)i * CDM * CDM, CDM, CDM);
            wconv(Wff1 + (size_t)i * CDM * CDM, Wff1T + (size_t)i * CDM * CDM, CDM, CDM);
            wconv(Wff2 + (size_t)i * CDM * CDM, Wff2T + (size_t)i * CDM * CDM, CDM, CDM);
        }

        // -------- embedding + graph prep --------
        tab_kernel<<<65, 128, 0, stream>>>(emb, tab_h);
        gather_kernel<<<(CN * CDIN) / 256, 256, 0, stream>>>(x_atoms, tab_h, x_h);
        fill_f32_kernel<<<(CN + 255) / 256, 256, 0, stream>>>(deg, 1.0f, CN);  // self loops
        deg_count_kernel<<<(CE + 255) / 256, 256, 0, stream>>>(ei, deg);
        dis_kernel<<<(CN + 255) / 256, 256, 0, stream>>>(deg, dis);

        // -------- GCN layer 1 (relu) --------
        gemm(x_h, Wg1T, nullptr, nullptr, xw, nullptr, CN, CDIN, CDIN, 0);
        fill_f32_kernel<<<(CN * CDIN) / 256, 256, 0, stream>>>(agg, 0.0f, CN * CDIN);
        edge_agg_kernel<<<((size_t)CE * CDIN) / 256, 256, 0, stream>>>(ei, dis, xw, agg);
        gcn_final_kernel<<<(CN * CDIN) / 256, 256, 0, stream>>>(agg, xw, dis, bg1, x_h, 1);

        // -------- GCN layer 2 --------
        gemm(x_h, Wg2T, nullptr, nullptr, xw, nullptr, CN, CDIN, CDIN, 0);
        fill_f32_kernel<<<(CN * CDIN) / 256, 256, 0, stream>>>(agg, 0.0f, CN * CDIN);
        edge_agg_kernel<<<((size_t)CE * CDIN) / 256, 256, 0, stream>>>(ei, dis, xw, agg);
        gcn_final_kernel<<<(CN * CDIN) / 256, 256, 0, stream>>>(agg, xw, dis, bg2, xg_h, 0);

        // -------- scatter into padded sequences, project to D_M --------
        fill_f16_kernel<<<(CT * CDIN) / 256, 256, 0, stream>>>(pad_h, CT * CDIN);
        scatter_kernel<<<(CN * CDIN) / 256, 256, 0, stream>>>(xg_h, pad_h);
        gemm(pad_h, WpT, bp, nullptr, h, h_h, CT, CDM, CDIN, 0);

        // -------- transformer layers --------
        for (int i = 0; i < CL; ++i) {
            const _Float16* WqT  = WqkvT + (size_t)i * 768 * CDM;
            const _Float16* WoTi = WoT   + (size_t)i * CDM * CDM;
            const _Float16* W1T  = Wff1T + (size_t)i * CDM * CDM;
            const _Float16* W2T  = Wff2T + (size_t)i * CDM * CDM;

            gemm(h_h, WqT, bqkv + (size_t)i * 768, nullptr, nullptr, qkv_h,
                 CT, 768, CDM, 0);
            attn_kernel<<<CB * CH, 64, 0, stream>>>(qkv_h, attn_h);
            gemm(attn_h, WoTi, bo + (size_t)i * CDM, h, tmp, nullptr,
                 CT, CDM, CDM, 0);
            ln_kernel<<<CT, 256, 0, stream>>>(tmp, ln1 + (size_t)i * 2 * CDM, h, h_h);
            gemm(h_h, W1T, bff1 + (size_t)i * CDM, nullptr, nullptr, ff_h,
                 CT, CDM, CDM, 1);
            gemm(ff_h, W2T, bff2 + (size_t)i * CDM, h, tmp, nullptr,
                 CT, CDM, CDM, 0);
            float* outF = (i == CL - 1) ? (float*)d_out : h;
            ln_kernel<<<CT, 256, 0, stream>>>(tmp, ln2 + (size_t)i * 2 * CDM, outF, h_h);
        }
    }
}
